// Encoder_69406671503579
// MI455X (gfx1250) — matrix operand order repaired and checked by reading, not compiled, not run on hardware
//
#include <hip/hip_runtime.h>
#include <hip/hip_bf16.h>

#define T_  6
#define NU_ 4096
#define NC_ 1536
#define DH_ 32
#define DO_ 16
#define SPLIT_ 4   // K-split factor for the big GEMMs (occupancy / latency hiding)

typedef __attribute__((ext_vector_type(16))) __bf16          v16bf;
typedef __attribute__((ext_vector_type(8)))  float           v8f;

// fp32 -> bf16 storage (used by the packing kernel)
__device__ __forceinline__ unsigned short f2bf(float f) {
    unsigned u = __float_as_uint(f);
    unsigned r = (u + 0x7FFFu + ((u >> 16) & 1u)) >> 16;
    return (unsigned short)r;
}

// ---- WMMA A-operand loader (16-bit A 16x32 layout, ISA 7.12.2) -------------
// lane L: M = L%16, g = L/16.  element e<8 -> K = g*8+e ; e>=8 -> K = 16+g*8+(e-8)
__device__ __forceinline__ v16bf load_A_dense(const float* rowp, int g) {
    const float4* p0 = (const float4*)(rowp + g * 8);
    const float4* p1 = (const float4*)(rowp + 16 + g * 8);
    float4 f0 = p0[0], f1 = p0[1];
    float4 f2 = p1[0], f3 = p1[1];
    v16bf a;
    a[0]  = (__bf16)f0.x; a[1]  = (__bf16)f0.y; a[2]  = (__bf16)f0.z; a[3]  = (__bf16)f0.w;
    a[4]  = (__bf16)f1.x; a[5]  = (__bf16)f1.y; a[6]  = (__bf16)f1.z; a[7]  = (__bf16)f1.w;
    a[8]  = (__bf16)f2.x; a[9]  = (__bf16)f2.y; a[10] = (__bf16)f2.z; a[11] = (__bf16)f2.w;
    a[12] = (__bf16)f3.x; a[13] = (__bf16)f3.y; a[14] = (__bf16)f3.z; a[15] = (__bf16)f3.w;
    return a;
}

// transposed A: element (m=col of W, k) read from row-major W[k][ldw]
__device__ __forceinline__ v16bf load_A_trans(const float* Wt, int ldw, int col,
                                              int k0, int g) {
    v16bf a;
#pragma unroll
    for (int e = 0; e < 16; ++e) {
        int k = k0 + ((e < 8) ? (g * 8 + e) : (16 + g * 8 + (e - 8)));
        a[e] = (__bf16)Wt[(size_t)k * ldw + col];
    }
    return a;
}

__device__ __forceinline__ v8f wmma_bf16(v16bf a, v16bf b, v8f c) {
    return __builtin_amdgcn_wmma_f32_16x16x32_bf16(
        false, a, false, b, (short)0, c, false, false);
}

// ---------------------------------------------------------------------------
// Kernel 1: feature projections X @ W[t]  ->  bf16, pre-swizzled B layout
//   B tile for (t,ktile,nblk): 512 bf16 = 32 lanes x 16 elems;
//   element for (kin,n): lane = ((kin/8)%2)*16 + n%16, elem = kin%8 + (kin/16)*8
// ---------------------------------------------------------------------------
__global__ __launch_bounds__(256) void k_xform(
    const float* __restrict__ X_U, const float* __restrict__ X_C,
    const float* __restrict__ Wuu, const float* __restrict__ Wuc,
    const float* __restrict__ Wcc, const float* __restrict__ Wcu,
    unsigned short* Buu, unsigned short* Buc,
    unsigned short* Bcc, unsigned short* Bcu) {
    long gid = (long)blockIdx.x * blockDim.x + threadIdx.x;
    const long s0 = (long)T_ * NU_ * 32;
    const long s1 = s0 + (long)T_ * NC_ * 32;
    const long s2 = s1 + (long)T_ * NC_ * 32;
    const long s3 = s2 + (long)T_ * NU_ * 32;
    const float* X; const float* W; unsigned short* Bp; int Krows; long r;
    if      (gid < s0) { r = gid;      X = X_U; W = Wuu; Bp = Buu; Krows = NU_; }
    else if (gid < s1) { r = gid - s0; X = X_C; W = Wuc; Bp = Buc; Krows = NC_; }
    else if (gid < s2) { r = gid - s1; X = X_C; W = Wcc; Bp = Bcc; Krows = NC_; }
    else if (gid < s3) { r = gid - s2; X = X_U; W = Wcu; Bp = Bcu; Krows = NU_; }
    else return;
    int  n   = (int)(r % 32);
    long rr  = r / 32;
    int  row = (int)(rr % Krows);
    int  t   = (int)(rr / Krows);
    const float* x = X + (size_t)row * 32;
    const float* w = W + (size_t)t * 32 * 32;
    float acc = 0.f;
#pragma unroll
    for (int f = 0; f < 32; ++f) acc += x[f] * w[f * 32 + n];
    int kt = row >> 5, kin = row & 31;
    int gg = (kin >> 3) & 1, e = (kin & 7) | ((kin >> 4) << 3), nblk = n >> 4, nin = n & 15;
    Bp[((size_t)(t * (Krows / 32) + kt) * 2 + nblk) * 512 + (gg * 16 + nin) * 16 + e]
        = f2bf(acc);
}

// ---------------------------------------------------------------------------
// Kernel 2: partial h_u[t] += w_uu[t] @ xu_u + w_uc[t] @ xc_u   [4096 x 32]
// one wave = one 16x32 output tile; K split SPLIT_ ways across gridDim.z;
// fp32 atomic accumulate into zeroed pre-activation buffer
// ---------------------------------------------------------------------------
__global__ __launch_bounds__(256) void k_hu(
    const float* __restrict__ w_uu, const float* __restrict__ w_uc,
    const unsigned short* __restrict__ Buu, const unsigned short* __restrict__ Buc,
    float* __restrict__ h_u) {
    const int t     = blockIdx.y;
    const int z     = blockIdx.z;
    const int wv    = threadIdx.x >> 5;
    const int lane  = threadIdx.x & 31;
    const int g     = lane >> 4;
    const int ml    = lane & 15;
    const int mbase = (blockIdx.x * 8 + wv) * 16;

    const float* W1 = w_uu + (size_t)t * NU_ * NU_ + (size_t)(mbase + ml) * NU_;
    const float* W2 = w_uc + (size_t)t * NU_ * NC_ + (size_t)(mbase + ml) * NC_;
    const unsigned short* B1 = Buu + (size_t)t * (NU_ / 32) * 2 * 512;
    const unsigned short* B2 = Buc + (size_t)t * (NC_ / 32) * 2 * 512;

    const int kt1b = z * (NU_ / 32 / SPLIT_), kt1e = kt1b + NU_ / 32 / SPLIT_;
    const int kt2b = z * (NC_ / 32 / SPLIT_), kt2e = kt2b + NC_ / 32 / SPLIT_;

    v8f c0 = {}, c1 = {};
    for (int kt = kt1b; kt < kt1e; ++kt) {
        __builtin_prefetch(W1 + kt * 32 + 128, 0, 3);
        v16bf a  = load_A_dense(W1 + kt * 32, g);
        v16bf b0 = *(const v16bf*)(B1 + ((size_t)kt * 2 + 0) * 512 + lane * 16);
        v16bf b1 = *(const v16bf*)(B1 + ((size_t)kt * 2 + 1) * 512 + lane * 16);
        c0 = wmma_bf16(a, b0, c0);
        c1 = wmma_bf16(a, b1, c1);
    }
    for (int kt = kt2b; kt < kt2e; ++kt) {
        __builtin_prefetch(W2 + kt * 32 + 128, 0, 3);
        v16bf a  = load_A_dense(W2 + kt * 32, g);
        v16bf b0 = *(const v16bf*)(B2 + ((size_t)kt * 2 + 0) * 512 + lane * 16);
        v16bf b1 = *(const v16bf*)(B2 + ((size_t)kt * 2 + 1) * 512 + lane * 16);
        c0 = wmma_bf16(a, b0, c0);
        c1 = wmma_bf16(a, b1, c1);
    }
#pragma unroll
    for (int r = 0; r < 8; ++r) {
        int row = mbase + r + 8 * g;
        atomicAdd(&h_u[((size_t)t * NU_ + row) * 32 + ml],      c0[r]);
        atomicAdd(&h_u[((size_t)t * NU_ + row) * 32 + 16 + ml], c1[r]);
    }
}

// ---------------------------------------------------------------------------
// Kernel 3: partial h_c[t] += w_cc[t] @ xc_c + w_uc[t]^T @ xu_c  [1536 x 32]
// ---------------------------------------------------------------------------
__global__ __launch_bounds__(256) void k_hc(
    const float* __restrict__ w_cc, const float* __restrict__ w_uc,
    const unsigned short* __restrict__ Bcc, const unsigned short* __restrict__ Bcu,
    float* __restrict__ h_c) {
    const int t     = blockIdx.y;
    const int z     = blockIdx.z;
    const int wv    = threadIdx.x >> 5;
    const int lane  = threadIdx.x & 31;
    const int g     = lane >> 4;
    const int ml    = lane & 15;
    const int mbase = (blockIdx.x * 8 + wv) * 16;

    const float* W1  = w_cc + (size_t)t * NC_ * NC_ + (size_t)(mbase + ml) * NC_;
    const float* W2t = w_uc + (size_t)t * NU_ * NC_;   // [NU_][NC_], used transposed
    const unsigned short* B1 = Bcc + (size_t)t * (NC_ / 32) * 2 * 512;
    const unsigned short* B2 = Bcu + (size_t)t * (NU_ / 32) * 2 * 512;

    const int kt1b = z * (NC_ / 32 / SPLIT_), kt1e = kt1b + NC_ / 32 / SPLIT_;
    const int kt2b = z * (NU_ / 32 / SPLIT_), kt2e = kt2b + NU_ / 32 / SPLIT_;

    v8f c0 = {}, c1 = {};
    for (int kt = kt1b; kt < kt1e; ++kt) {
        __builtin_prefetch(W1 + kt * 32 + 128, 0, 3);
        v16bf a  = load_A_dense(W1 + kt * 32, g);
        v16bf b0 = *(const v16bf*)(B1 + ((size_t)kt * 2 + 0) * 512 + lane * 16);
        v16bf b1 = *(const v16bf*)(B1 + ((size_t)kt * 2 + 1) * 512 + lane * 16);
        c0 = wmma_bf16(a, b0, c0);
        c1 = wmma_bf16(a, b1, c1);
    }
    for (int kt = kt2b; kt < kt2e; ++kt) {
        v16bf a  = load_A_trans(W2t, NC_, mbase + ml, kt * 32, g);
        v16bf b0 = *(const v16bf*)(B2 + ((size_t)kt * 2 + 0) * 512 + lane * 16);
        v16bf b1 = *(const v16bf*)(B2 + ((size_t)kt * 2 + 1) * 512 + lane * 16);
        c0 = wmma_bf16(a, b0, c0);
        c1 = wmma_bf16(a, b1, c1);
    }
#pragma unroll
    for (int r = 0; r < 8; ++r) {
        int row = mbase + r + 8 * g;
        atomicAdd(&h_c[((size_t)t * NC_ + row) * 32 + ml],      c0[r]);
        atomicAdd(&h_c[((size_t)t * NC_ + row) * 32 + 16 + ml], c1[r]);
    }
}

// ---------------------------------------------------------------------------
// Kernel 4: ReLU in place + per-(t,col) sums (LDS-reduced before global atomics)
// block = 256 threads = 8 rows of one (buffer, t); col = tid % 32
// ---------------------------------------------------------------------------
__global__ __launch_bounds__(256) void k_finish(
    float* __restrict__ h_u, float* __restrict__ h_c,
    float* __restrict__ hsum_u, float* __restrict__ hsum_c) {
    __shared__ float red[32];
    long idx = (long)blockIdx.x * 256 + threadIdx.x;
    const long nU = (long)T_ * NU_ * 32;
    const long nC = (long)T_ * NC_ * 32;
    float* h; float* hsum; int t; long i;
    if (idx < nU)           { h = h_u; hsum = hsum_u; i = idx;      t = (int)(i / ((long)NU_ * 32)); }
    else if (idx < nU + nC) { h = h_c; hsum = hsum_c; i = idx - nU; t = (int)(i / ((long)NC_ * 32)); }
    else return;
    if (threadIdx.x < 32) red[threadIdx.x] = 0.f;
    __syncthreads();
    float v = h[i];
    v = v > 0.f ? v : 0.f;
    h[i] = v;
    atomicAdd(&red[threadIdx.x & 31], v);   // ds_add_f32, bank-friendly
    __syncthreads();
    if (threadIdx.x < 32) atomicAdd(&hsum[t * 32 + threadIdx.x], red[threadIdx.x]);
}

// ---------------------------------------------------------------------------
// Kernel 5: attention alphas.  alpha[0..5]=alpha_u (ctx=h_c), [6..11]=alpha_c
// ---------------------------------------------------------------------------
__global__ void k_attn(const float* __restrict__ hsum_u,
                       const float* __restrict__ hsum_c,
                       const float* __restrict__ Wa, const float* __restrict__ va,
                       float* __restrict__ alpha) {
    __shared__ float sc[12];
    int tid = threadIdx.x;
    if (tid < 12) {
        int which = tid / 6, t = tid % 6;
        const float* hs = (which == 0) ? hsum_c : hsum_u;
        float nn = (which == 0) ? (float)NC_ : (float)NU_;
        float s = 0.f;
        for (int j = 0; j < 32; ++j) {
            float dot = 0.f;
            for (int h = 0; h < 32; ++h) dot += (hs[t * 32 + h] / nn) * Wa[h * 32 + j];
            s += tanhf(dot) * va[j];
        }
        sc[tid] = s;
    }
    __syncthreads();
    if (tid < 2) {
        float mx = -1e30f;
        for (int t = 0; t < 6; ++t) mx = fmaxf(mx, sc[tid * 6 + t]);
        float ex[6], den = 0.f;
        for (int t = 0; t < 6; ++t) { ex[t] = expf(sc[tid * 6 + t] - mx); den += ex[t]; }
        for (int t = 0; t < 6; ++t) alpha[tid * 6 + t] = ex[t] / den;
    }
}

// ---------------------------------------------------------------------------
// Kernel 6: per node: y[t] = h[t]@Wo[t]; out = sum_t alpha[t]*y[t] @ Wm_t + bm
// ---------------------------------------------------------------------------
__global__ __launch_bounds__(256) void k_post(
    const float* __restrict__ h_u, const float* __restrict__ h_c,
    const float* __restrict__ Wou, const float* __restrict__ Woc,
    const float* __restrict__ Wm, const float* __restrict__ bm,
    const float* __restrict__ alpha, float* __restrict__ out) {
    int idx = blockIdx.x * blockDim.x + threadIdx.x;
    const float *h, *Wo, *al; float* o; int N, i;
    if (idx < NU_)              { h = h_u; Wo = Wou; al = alpha;     o = out;                      N = NU_; i = idx; }
    else if (idx < NU_ + NC_)   { h = h_c; Wo = Woc; al = alpha + 6; o = out + (size_t)NU_ * DO_;  N = NC_; i = idx - NU_; }
    else return;

    float acc[DO_];
#pragma unroll
    for (int o2 = 0; o2 < DO_; ++o2) acc[o2] = bm[o2];

    for (int t = 0; t < T_; ++t) {
        const float* hr = h + ((size_t)t * N + i) * 32;
        float hreg[32];
#pragma unroll
        for (int hh = 0; hh < 32; ++hh) hreg[hh] = hr[hh];
        float at = al[t];
#pragma unroll
        for (int oo = 0; oo < DO_; ++oo) {
            float y = 0.f;
#pragma unroll
            for (int hh = 0; hh < 32; ++hh) y += hreg[hh] * Wo[((size_t)t * 32 + hh) * DO_ + oo];
            float ya = at * y;
#pragma unroll
            for (int o2 = 0; o2 < DO_; ++o2) acc[o2] += ya * Wm[(t * DO_ + oo) * DO_ + o2];
        }
    }
#pragma unroll
    for (int o2 = 0; o2 < DO_; ++o2) o[(size_t)i * DO_ + o2] = acc[o2];
}

// ---------------------------------------------------------------------------
extern "C" void kernel_launch(void* const* d_in, const int* in_sizes, int n_in,
                              void* d_out, int out_size, void* d_ws, size_t ws_size,
                              hipStream_t stream) {
    const float* w_uu = (const float*)d_in[0];
    const float* w_uc = (const float*)d_in[1];
    const float* w_cc = (const float*)d_in[2];
    const float* X_U  = (const float*)d_in[3];
    const float* X_C  = (const float*)d_in[4];
    const float* Wuu  = (const float*)d_in[5];
    const float* Wuc  = (const float*)d_in[6];
    const float* Wcc  = (const float*)d_in[7];
    const float* Wcu  = (const float*)d_in[8];
    const float* Wou  = (const float*)d_in[9];
    const float* Woc  = (const float*)d_in[10];
    const float* Wa   = (const float*)d_in[11];
    const float* va   = (const float*)d_in[12];
    const float* Wm   = (const float*)d_in[13];
    const float* bm   = (const float*)d_in[14];
    float* out = (float*)d_out;

    // workspace layout (256B aligned slabs)
    auto al256 = [](size_t x) { return (x + 255) & ~(size_t)255; };
    char* ws = (char*)d_ws;
    size_t off = 0;
    const size_t sz_Buu = (size_t)T_ * (NU_ / 32) * 2 * 512 * 2;
    const size_t sz_Buc = (size_t)T_ * (NC_ / 32) * 2 * 512 * 2;
    const size_t sz_hu  = (size_t)T_ * NU_ * 32 * 4;   // multiple of 256
    const size_t sz_hc  = (size_t)T_ * NC_ * 32 * 4;   // multiple of 256
    unsigned short* Buu = (unsigned short*)(ws + off); off = al256(off + sz_Buu);
    unsigned short* Buc = (unsigned short*)(ws + off); off = al256(off + sz_Buc);
    unsigned short* Bcc = (unsigned short*)(ws + off); off = al256(off + sz_Buc);
    unsigned short* Bcu = (unsigned short*)(ws + off); off = al256(off + sz_Buu);
    float* h_u    = (float*)(ws + off); off += sz_hu;          // contiguous with h_c
    float* h_c    = (float*)(ws + off); off = al256(off + sz_hc);
    float* hsum_u = (float*)(ws + off); off = al256(off + (size_t)T_ * 32 * 4 * 2);
    float* hsum_c = hsum_u + T_ * 32;
    float* alpha  = (float*)(ws + off); off = al256(off + 12 * 4);
    (void)ws_size; (void)out_size; (void)n_in; (void)in_sizes;

    // zero accumulators: h_u+h_c (contiguous) and hsum_u+hsum_c (contiguous)
    hipMemsetAsync(h_u, 0, sz_hu + sz_hc, stream);
    hipMemsetAsync(hsum_u, 0, (size_t)T_ * 32 * 4 * 2, stream);

    // 1) feature projections -> pre-swizzled bf16 B operands
    long total = (long)T_ * 32 * (NU_ + NC_ + NC_ + NU_);
    k_xform<<<dim3((unsigned)((total + 255) / 256)), 256, 0, stream>>>(
        X_U, X_C, Wuu, Wuc, Wcc, Wcu, Buu, Buc, Bcc, Bcu);

    // 2) big WMMA GEMMs, K-split for occupancy (memory-bound adjacency streams)
    k_hu<<<dim3(NU_ / 128, T_, SPLIT_), 256, 0, stream>>>(w_uu, w_uc, Buu, Buc, h_u);
    k_hc<<<dim3(NC_ / 128, T_, SPLIT_), 256, 0, stream>>>(w_cc, w_uc, Bcc, Bcu, h_c);

    // 3) ReLU + column sums (LDS-reduced)
    long nfin = (long)T_ * (NU_ + NC_) * 32;
    k_finish<<<dim3((unsigned)((nfin + 255) / 256)), 256, 0, stream>>>(
        h_u, h_c, hsum_u, hsum_c);

    // 4) attention alphas
    k_attn<<<1, 64, 0, stream>>>(hsum_u, hsum_c, Wa, va, alpha);

    // 5) per-node output head
    k_post<<<dim3((NU_ + NC_ + 255) / 256), 256, 0, stream>>>(
        h_u, h_c, Wou, Woc, Wm, bm, alpha, out);
}
